// GINConv_12137577578700
// MI455X (gfx1250) — matrix operand order repaired and checked by reading, not compiled, not run on hardware
//
#include <hip/hip_runtime.h>
#include <hip/hip_bf16.h>

// ---------------------------------------------------------------------------
// CDNA5 (gfx1250) fused GINConv:
//   agg = x + scatter_add(x[col] + emb1[a0] + emb2[a1])        (edge kernel)
//   out = relu(agg @ W1 + b1) @ W2 + b2                        (fused WMMA MLP)
//
// MLP: one workgroup = 64 nodes (4 M-sub-tiles of 16). Each wave loads its
// B-fragments (pre-transposed f16 weights) ONCE per k-step and reuses them
// across 4 A-fragments -> 64 v_wmma per wave, 4x fewer weight fetches.
// ---------------------------------------------------------------------------

typedef __attribute__((ext_vector_type(16))) _Float16 v16h;
typedef __attribute__((ext_vector_type(8)))  _Float16 v8h_t;
typedef __attribute__((ext_vector_type(8)))  float    v8f;

union F16x16 { v16h v; v8h_t h[2]; };

#define D_IN     128
#define D_HID    256
#define M_TILE   64    // rows per workgroup
#define SA       136   // lds_a row stride in halves (128 + 8 pad -> conflict-free)
#define SH       264   // lds_h row stride in halves (256 + 8 pad)

// ---------------------------------------------------------------------------
// Kernel 1: agg = x   (covers the "+x" self term; agg may alias d_out)
// ---------------------------------------------------------------------------
__global__ void init_agg_kernel(const float* __restrict__ x, float* agg, int n4) {
    int i = blockIdx.x * blockDim.x + threadIdx.x;
    if (i < n4) ((float4*)agg)[i] = ((const float4*)x)[i];
}

// ---------------------------------------------------------------------------
// Kernel 2: transpose + fp16-quantize weights: WT[n*K + k] = (f16) W[k*N + n]
// ---------------------------------------------------------------------------
__global__ void transpose_f16_kernel(const float* __restrict__ W, _Float16* WT,
                                     int K, int N) {
    int idx = blockIdx.x * blockDim.x + threadIdx.x;
    if (idx >= K * N) return;
    int n = idx / K;
    int k = idx - n * K;
    WT[idx] = (_Float16)W[k * N + n];
}

// ---------------------------------------------------------------------------
// Kernel 3: edge messages + scatter-add.  One wave32 per edge, 4 floats/lane.
// Edge indices are wave-uniform -> compiler scalarizes the index loads.
// ---------------------------------------------------------------------------
__global__ void edge_kernel(const float* __restrict__ x,
                            const int*   __restrict__ ei,   // [2*E]
                            const int*   __restrict__ ea,   // [E][2]
                            const float* __restrict__ emb1, // [5][128]
                            const float* __restrict__ emb2, // [3][128]
                            float* agg, int E) {
    int gid  = blockIdx.x * blockDim.x + threadIdx.x;
    int e    = gid >> 5;
    int lane = gid & 31;
    if (e >= E) return;

    int dst = ei[e];          // row: destination node
    int src = ei[E + e];      // col: neighbor node
    int a0  = ea[2 * e + 0];
    int a1  = ea[2 * e + 1];

    float4 m  = ((const float4*)(x    + (size_t)src * D_IN))[lane];
    float4 e1 = ((const float4*)(emb1 + (size_t)a0  * D_IN))[lane];
    float4 e2 = ((const float4*)(emb2 + (size_t)a1  * D_IN))[lane];
    m.x += e1.x + e2.x;
    m.y += e1.y + e2.y;
    m.z += e1.z + e2.z;
    m.w += e1.w + e2.w;

    float* p = agg + (size_t)dst * D_IN + 4 * lane;
    atomicAdd(p + 0, m.x);
    atomicAdd(p + 1, m.y);
    atomicAdd(p + 2, m.z);
    atomicAdd(p + 3, m.w);
}

// ---------------------------------------------------------------------------
// WMMA fragment loaders (layouts per CDNA5 ISA §7.12.2)
// ---------------------------------------------------------------------------
// A (16xK f16): lane holds row (lane&15); lanes 0-15 get K = kb+0..7 / kb+16..23,
// lanes 16-31 get K = kb+8..15 / kb+24..31.
__device__ __forceinline__ v16h load_a_frag(const _Float16* a, int stride,
                                            int kbase, int lane) {
    int row = lane & 15;
    int kh  = (lane & 16) ? 8 : 0;
    const _Float16* p = a + row * stride + kbase + kh;
    F16x16 u;
    u.h[0] = *(const v8h_t*)(p);        // K = kb+kh   .. kb+kh+7
    u.h[1] = *(const v8h_t*)(p + 16);   // K = kb+16+kh.. kb+16+kh+7
    return u.v;
}

// B (Kx16 f16) from K-major (pre-transposed) weights WT[n][K]:
// lane holds column (lane&15); lanes 0-15 get K = kb+0..7 / kb+16..23, lanes 16-31 K = kb+8..15 / kb+24..31.
__device__ __forceinline__ v16h load_b_frag(const _Float16* __restrict__ WT, int ldk,
                                            int nbase, int kbase, int lane) {
    int n  = nbase + (lane & 15);
    int kb = kbase + ((lane & 16) ? 8 : 0);
    const _Float16* p = WT + (size_t)n * ldk + kb;
    F16x16 u;
    u.h[0] = *(const v8h_t*)(p);
    u.h[1] = *(const v8h_t*)(p + 16);
    return u.v;
}

__device__ __forceinline__ v8f wmma_f16(v16h a, v16h b, v8f c) {
    return __builtin_amdgcn_wmma_f32_16x16x32_f16(false, a, false, b,
                                                  (short)0, c, false, false);
}

// ---------------------------------------------------------------------------
// Kernel 4: fused MLP.  256 threads (8 wave32) per workgroup, 64 nodes each.
//   GEMM1: (64x128) @ (128x256) -> relu -> LDS (f16)
//   GEMM2: (64x256) @ (256x128) -> +b2 -> out
// Each wave: GEMM1 covers 2 N-tiles x 4 M-tiles, GEMM2 1 N-tile x 4 M-tiles.
// ---------------------------------------------------------------------------
__global__ void __launch_bounds__(256)
mlp_fused_kernel(const float* agg,
                 const _Float16* __restrict__ WT1,  // [256][128] f16 (K-major)
                 const float*    __restrict__ b1,   // [256]
                 const _Float16* __restrict__ WT2,  // [128][256] f16 (K-major)
                 const float*    __restrict__ b2,   // [128]
                 float* out, int nNodes) {
    __shared__ _Float16 lds_a[M_TILE * SA];   // 17408 B
    __shared__ _Float16 lds_h[M_TILE * SH];   // 33792 B

    const int tid  = threadIdx.x;
    const int lane = tid & 31;
    const int w    = tid >> 5;                // wave id 0..7
    const int m0   = blockIdx.x * M_TILE;

    // ---- stage agg rows (fp32 -> f16) into LDS: 64 rows x 128 cols,
    //      2048 float4 total -> 8 float4 per thread, coalesced.
    {
#pragma unroll
        for (int j = 0; j < 8; ++j) {
            int i4   = tid + 256 * j;        // 0..2047
            int row  = i4 >> 5;              // 32 float4 per row
            int col4 = i4 & 31;
            int m = m0 + row;
            if (m >= nNodes) m = nNodes - 1; // clamp (grid divides exactly anyway)
            float4 p = ((const float4*)(agg + (size_t)m * D_IN))[col4];
            _Float16* dp = lds_a + row * SA + col4 * 4;
            dp[0] = (_Float16)p.x;
            dp[1] = (_Float16)p.y;
            dp[2] = (_Float16)p.z;
            dp[3] = (_Float16)p.w;
        }
    }
    __syncthreads();

    // ---- GEMM1: wave covers N-tiles nb0, nb1; B-frags reused over 4 M-tiles
    const int nb0 = w * 32;
    const int nb1 = w * 32 + 16;
    v8f c[4][2] = {};
#pragma unroll
    for (int kk = 0; kk < 4; ++kk) {              // K = 128 = 4 x 32
        int kb = kk * 32;
        v16h bf0 = load_b_frag(WT1, 128, nb0, kb, lane);
        v16h bf1 = load_b_frag(WT1, 128, nb1, kb, lane);
#pragma unroll
        for (int mt = 0; mt < 4; ++mt) {
            v16h af = load_a_frag(lds_a + mt * 16 * SA, SA, kb, lane);
            c[mt][0] = wmma_f16(af, bf0, c[mt][0]);
            c[mt][1] = wmma_f16(af, bf1, c[mt][1]);
        }
    }

    // ---- bias + ReLU, f16, into LDS  (C/D layout: n = lane&15, rows mb+0..7)
    {
        int n  = lane & 15;
        int mb = (lane & 16) ? 8 : 0;
        float bias0 = b1[nb0 + n];
        float bias1 = b1[nb1 + n];
#pragma unroll
        for (int mt = 0; mt < 4; ++mt) {
#pragma unroll
            for (int r = 0; r < 8; ++r) {
                float h0 = c[mt][0][r] + bias0; h0 = h0 > 0.f ? h0 : 0.f;
                float h1 = c[mt][1][r] + bias1; h1 = h1 > 0.f ? h1 : 0.f;
                int rowh = (mt * 16 + mb + r) * SH;
                lds_h[rowh + nb0 + n] = (_Float16)h0;
                lds_h[rowh + nb1 + n] = (_Float16)h1;
            }
        }
    }
    __syncthreads();

    // ---- GEMM2: wave covers 1 N-tile; B-frag reused over 4 M-tiles
    const int nb = w * 16;
    v8f d[4] = {};
#pragma unroll
    for (int kk = 0; kk < 8; ++kk) {              // K = 256 = 8 x 32
        int kb = kk * 32;
        v16h bf = load_b_frag(WT2, 256, nb, kb, lane);
#pragma unroll
        for (int mt = 0; mt < 4; ++mt) {
            v16h af = load_a_frag(lds_h + mt * 16 * SH, SH, kb, lane);
            d[mt] = wmma_f16(af, bf, d[mt]);
        }
    }

    // ---- + b2, store fp32
    {
        int n  = lane & 15;
        int mb = (lane & 16) ? 8 : 0;
        float bias = b2[nb + n];
#pragma unroll
        for (int mt = 0; mt < 4; ++mt) {
#pragma unroll
            for (int r = 0; r < 8; ++r) {
                int m = m0 + mt * 16 + mb + r;
                if (m < nNodes) out[(size_t)m * D_IN + nb + n] = d[mt][r] + bias;
            }
        }
    }
}

// ---------------------------------------------------------------------------
// Host launcher
// ---------------------------------------------------------------------------
extern "C" void kernel_launch(void* const* d_in, const int* in_sizes, int n_in,
                              void* d_out, int out_size, void* d_ws, size_t ws_size,
                              hipStream_t stream) {
    const float* x    = (const float*)d_in[0];   // [N,128]
    const int*   ei   = (const int*)  d_in[1];   // [2,E]
    const int*   ea   = (const int*)  d_in[2];   // [E,2]
    const float* W1   = (const float*)d_in[3];   // [128,256]
    const float* b1   = (const float*)d_in[4];   // [256]
    const float* W2   = (const float*)d_in[5];   // [256,128]
    const float* b2   = (const float*)d_in[6];   // [128]
    const float* emb1 = (const float*)d_in[7];   // [5,128]
    const float* emb2 = (const float*)d_in[8];   // [3,128]
    float* out = (float*)d_out;

    const int N = in_sizes[0] / D_IN;
    const int E = in_sizes[1] / 2;

    // Workspace layout: [WT1 f16 256*128][WT2 f16 128*256][optional agg f32]
    _Float16* WT1 = (_Float16*)d_ws;
    _Float16* WT2 = WT1 + D_HID * D_IN;
    size_t wt_bytes  = (size_t)2 * D_HID * D_IN * sizeof(_Float16);
    size_t agg_bytes = (size_t)N * D_IN * sizeof(float);

    float* agg;
    if (ws_size >= wt_bytes + 256 + agg_bytes) {
        // agg in workspace (aligned up to 256B past the weights)
        agg = (float*)((char*)d_ws + ((wt_bytes + 255) & ~(size_t)255));
    } else {
        // alias d_out: each MLP workgroup reads its 64 rows of agg before
        // writing the same 64 rows of out -> safe under stream ordering.
        agg = out;
    }

    // 1) agg = x
    {
        int n4 = N * (D_IN / 4);
        init_agg_kernel<<<(n4 + 255) / 256, 256, 0, stream>>>(x, agg, n4);
    }
    // 2) pre-transpose + quantize weights to f16 (K-major for B-fragments)
    transpose_f16_kernel<<<(D_IN * D_HID + 255) / 256, 256, 0, stream>>>(W1, WT1, D_IN, D_HID);
    transpose_f16_kernel<<<(D_HID * D_IN + 255) / 256, 256, 0, stream>>>(W2, WT2, D_HID, D_IN);
    // 3) edge messages + scatter-add (1 wave per edge)
    {
        long long threads = (long long)E * 32;
        int blocks = (int)((threads + 255) / 256);
        edge_kernel<<<blocks, 256, 0, stream>>>(x, ei, ea, emb1, emb2, agg, E);
    }
    // 4) fused WMMA MLP (64 nodes per workgroup)
    {
        int blocks = (N + M_TILE - 1) / M_TILE;
        mlp_fused_kernel<<<blocks, 256, 0, stream>>>(agg, WT1, b1, WT2, b2, out, N);
    }
}
